// CGC_60249801228326
// MI455X (gfx1250) — matrix operand order repaired and checked by reading, not compiled, not run on hardware
//
#include <hip/hip_runtime.h>

typedef __attribute__((ext_vector_type(16))) __bf16 v16bf;
typedef __attribute__((ext_vector_type(8)))  __bf16 v8bf;
typedef __attribute__((ext_vector_type(8)))  float  v8f;

#define B_ 4096
#define I_ 1024
#define H_ 2048
#define O_ 512
#define LDH (H_ + 16)   // padded LDS row stride (bf16 elems): +32B rotates banks by 8

// ---------- helpers ----------

__device__ __forceinline__ unsigned short f2bf(float f) {
  unsigned u = __float_as_uint(f);
  u += 0x7FFFu + ((u >> 16) & 1u);      // round-to-nearest-even
  return (unsigned short)(u >> 16);
}

// A-matrix fragment, 16x32 bf16 (MxK), row-major source with leading dim `ld`.
// ISA layout: lanes 0-15: M=lane, elems0-7 = K0..7, elems8-15 = K16..23
//             lanes16-31: M=lane-16, elems0-7 = K8..15, elems8-15 = K24..31
__device__ __forceinline__ v16bf frag_a(const unsigned short* base, int ld,
                                        int row0, int k0, int lane) {
  const unsigned short* p =
      base + (size_t)(row0 + (lane & 15)) * ld + k0 + ((lane >> 4) << 3);
  v8bf lo = *reinterpret_cast<const v8bf*>(p);
  v8bf hi = *reinterpret_cast<const v8bf*>(p + 16);
  return __builtin_shufflevector(lo, hi, 0,1,2,3,4,5,6,7,8,9,10,11,12,13,14,15);
}

// B-matrix fragment, 32x16 bf16 (KxN), sourced from B^T stored row-major (N x K).
// ISA layout: lanes 0-15: N=lane, elems0-7 = K0..7, elems8-15 = K16..23
//             lanes16-31: N=lane-16, elems0-7 = K8..15, elems8-15 = K24..31
__device__ __forceinline__ v16bf frag_b(const unsigned short* baseT, int ld,
                                        int n0, int k0, int lane) {
  const unsigned short* p =
      baseT + (size_t)(n0 + (lane & 15)) * ld + k0 + ((lane >> 4) << 3);
  v8bf lo = *reinterpret_cast<const v8bf*>(p);
  v8bf hi = *reinterpret_cast<const v8bf*>(p + 16);
  return __builtin_shufflevector(lo, hi, 0,1,2,3,4,5,6,7,8,9,10,11,12,13,14,15);
}

__device__ __forceinline__ v8f wmma_bf16(v16bf a, v16bf b, v8f c) {
  return __builtin_amdgcn_wmma_f32_16x16x32_bf16(false, a, false, b,
                                                 (short)0, c, false, false);
}

// ---------- pre-pass kernels ----------

__global__ __launch_bounds__(256) void cgc_cvt_x_kernel(
    const float* __restrict__ x_sh, const float* __restrict__ x_t1,
    const float* __restrict__ x_t2, unsigned short* __restrict__ xb) {
  const size_t n = (size_t)B_ * I_;
  size_t i = (size_t)blockIdx.x * blockDim.x + threadIdx.x;
  if (i >= n) return;
  const float* src = (blockIdx.y == 0) ? x_sh : (blockIdx.y == 1) ? x_t1 : x_t2;
  xb[(size_t)blockIdx.y * n + i] = f2bf(src[i]);
}

// [slice, R, C] f32  ->  [slice, C, R] bf16   (LDS-tiled 32x32 transpose)
__global__ __launch_bounds__(256) void cgc_transpose_bf16_kernel(
    const float* __restrict__ in, unsigned short* __restrict__ outT,
    int R, int C) {
  __shared__ float tile[32][33];
  const size_t slice = (size_t)blockIdx.z * R * C;
  const int c0 = blockIdx.x * 32, r0 = blockIdx.y * 32;
  const int tx = threadIdx.x & 31, ty = threadIdx.x >> 5;   // 32 x 8
  #pragma unroll
  for (int k = 0; k < 4; ++k) {
    int r = r0 + ty + k * 8;
    tile[ty + k * 8][tx] = in[slice + (size_t)r * C + c0 + tx];
  }
  __syncthreads();
  #pragma unroll
  for (int k = 0; k < 4; ++k) {
    int c = c0 + ty + k * 8;
    outT[slice + (size_t)c * R + r0 + tx] = f2bf(tile[tx][ty + k * 8]);
  }
}

// softmax gates: gs [B,12], g1 [B,8], g2 [B,8]; one block (128 thr) per row
__global__ __launch_bounds__(128) void cgc_gate_kernel(
    const float* __restrict__ x_sh, const float* __restrict__ x_t1,
    const float* __restrict__ x_t2,
    const float* __restrict__ wg_sh, const float* __restrict__ bg_sh,
    const float* __restrict__ wg_t1, const float* __restrict__ bg_t1,
    const float* __restrict__ wg_t2, const float* __restrict__ bg_t2,
    float* __restrict__ gs, float* __restrict__ g1, float* __restrict__ g2) {
  __shared__ float red[12][128];
  const int b = blockIdx.x, t = threadIdx.x;
  const float* xs[3]  = {x_sh, x_t1, x_t2};
  const float* wgs[3] = {wg_sh, wg_t1, wg_t2};
  const float* bgs[3] = {bg_sh, bg_t1, bg_t2};
  float* outs[3]      = {gs, g1, g2};
  const int ncols[3]  = {12, 8, 8};
  for (int s = 0; s < 3; ++s) {
    const int nc = ncols[s];
    float acc[12];
    for (int c = 0; c < 12; ++c) acc[c] = 0.f;
    const float* x  = xs[s] + (size_t)b * I_;
    const float* wg = wgs[s];
    for (int i = t; i < I_; i += 128) {
      float xv = x[i];
      for (int c = 0; c < nc; ++c) acc[c] += xv * wg[i * nc + c];
    }
    for (int c = 0; c < nc; ++c) red[c][t] = acc[c];
    __syncthreads();
    if (t == 0) {
      float lg[12], mx = -3.4e38f;
      for (int c = 0; c < nc; ++c) {
        float sum = bgs[s][c];
        for (int k = 0; k < 128; ++k) sum += red[c][k];
        lg[c] = sum;
        if (sum > mx) mx = sum;
      }
      float den = 0.f;
      for (int c = 0; c < nc; ++c) { lg[c] = __expf(lg[c] - mx); den += lg[c]; }
      float inv = 1.f / den;
      for (int c = 0; c < nc; ++c) outs[s][(size_t)b * nc + c] = lg[c] * inv;
    }
    __syncthreads();
  }
}

// ---------- fused expert kernel ----------
// grid = (B/16, 3 groups).  Per block: 16 batch rows, the group's 4 experts:
//   h = relu(x@w1+b1) (bf16, LDS) -> o = h@w2+b2 -> gate-weighted LDS fp32 acc
// Final: one atomicAdd per output element per (block,target).
__global__ __launch_bounds__(256) void cgc_expert_kernel(
    const unsigned short* __restrict__ xb,    // [3][B][I] bf16 : sh,t1,t2
    const unsigned short* __restrict__ w1T,   // [12][H][I] bf16 : t1,t2,sh
    const unsigned short* __restrict__ w2T,   // [12][O][H] bf16
    const float* __restrict__ b1_t1, const float* __restrict__ b1_t2,
    const float* __restrict__ b1_sh,
    const float* __restrict__ b2_t1, const float* __restrict__ b2_t2,
    const float* __restrict__ b2_sh,
    const float* __restrict__ gs, const float* __restrict__ g1,
    const float* __restrict__ g2,
    float* __restrict__ out)                  // [3][B][O] : sh, task1, task2
{
  extern __shared__ char smem[];
  unsigned short* h = (unsigned short*)smem;                         // 16*LDH bf16
  float* outacc = (float*)(smem + (size_t)16 * LDH * 2);             // 3*16*O_ f32
  float* gw     = (float*)(smem + (size_t)16 * LDH * 2 + (size_t)3 * 16 * O_ * 4);

  const int tid   = threadIdx.x;
  const int lane  = tid & 31;
  const int wave  = tid >> 5;
  const int bBase = blockIdx.x * 16;
  const int grp   = blockIdx.y;          // 0=t1, 1=t2, 2=shared
  const int nt    = (grp == 2) ? 3 : 2;  // #output targets

  const unsigned short* x = xb + (size_t)((grp == 2) ? 0 : (grp + 1)) * B_ * I_;
  const float* b1g = (grp == 0) ? b1_t1 : (grp == 1) ? b1_t2 : b1_sh;
  const float* b2g = (grp == 0) ? b2_t1 : (grp == 1) ? b2_t2 : b2_sh;

  for (int i = tid; i < 3 * 16 * O_; i += 256) outacc[i] = 0.f;

  const int mo = (lane >> 4) << 3;   // C/D: VGPR r -> M = mo + r
  const int nn = lane & 15;          // C/D: N = lane & 15

  for (int je = 0; je < 4; ++je) {
    const int ge = grp * 4 + je;
    const unsigned short* w1e = w1T + (size_t)ge * H_ * I_;
    const unsigned short* w2e = w2T + (size_t)ge * O_ * H_;
    const float* b1 = b1g + je * H_;
    const float* b2 = b2g + je * O_;

    __syncthreads();   // previous expert's phase-2 reads of h / gw are done
    if (tid < 16) {
      int b = bBase + tid;
      gw[tid] = gs[b * 12 + ge];                           // out_sh weight
      if (grp == 0)      gw[16 + tid] = g1[b * 8 + je];
      else if (grp == 1) gw[16 + tid] = g2[b * 8 + je];
      else { gw[16 + tid] = g1[b * 8 + 4 + je];
             gw[32 + tid] = g2[b * 8 + 4 + je]; }
    }

    // phase 1: h[16 x 2048] = relu(x_tile @ w1 + b1) -> bf16 LDS
    for (int nT = wave; nT < H_ / 16; nT += 8) {
      const int n0 = nT * 16;
      v8f acc = {};
      for (int k = 0; k < I_; k += 32) {
        v16bf a = frag_a(x, I_, bBase, k, lane);
        v16bf b = frag_b(w1e, I_, n0, k, lane);
        acc = wmma_bf16(a, b, acc);
      }
      float bias = b1[n0 + nn];
      #pragma unroll
      for (int r = 0; r < 8; ++r) {
        float v = acc[r] + bias;
        h[(size_t)(mo + r) * LDH + n0 + nn] = f2bf(v > 0.f ? v : 0.f);
      }
    }
    __syncthreads();

    // phase 2: o[16 x 512] = h @ w2 + b2; gate-weighted accumulate (lane-owned)
    for (int nT = wave; nT < O_ / 16; nT += 8) {
      const int n0 = nT * 16;
      v8f acc = {};
      for (int k = 0; k < H_; k += 32) {
        v16bf a = frag_a(h, LDH, 0, k, lane);
        v16bf b = frag_b(w2e, H_, n0, k, lane);
        acc = wmma_bf16(a, b, acc);
      }
      float bias = b2[n0 + nn];
      #pragma unroll
      for (int r = 0; r < 8; ++r) {
        int m = mo + r;
        float v = acc[r] + bias;
        outacc[(size_t)m * O_ + n0 + nn]            += gw[m]      * v;
        outacc[16 * O_ + (size_t)m * O_ + n0 + nn]  += gw[16 + m] * v;
        if (grp == 2)
          outacc[2 * 16 * O_ + (size_t)m * O_ + n0 + nn] += gw[32 + m] * v;
      }
    }
  }
  __syncthreads();

  const size_t off1 = (grp == 1) ? (size_t)2 * B_ * O_ : (size_t)B_ * O_;
  for (int idx = tid; idx < nt * 16 * O_; idx += 256) {
    int t   = idx / (16 * O_);
    int rem = idx - t * (16 * O_);
    int row = rem / O_;
    int col = rem - row * O_;
    size_t toff = (t == 0) ? 0 : (t == 1) ? off1 : (size_t)2 * B_ * O_;
    atomicAdd(out + toff + (size_t)(bBase + row) * O_ + col, outacc[idx]);
  }
}

// ---------- host ----------

extern "C" void kernel_launch(void* const* d_in, const int* in_sizes, int n_in,
                              void* d_out, int out_size, void* d_ws, size_t ws_size,
                              hipStream_t stream) {
  (void)in_sizes; (void)n_in; (void)out_size; (void)ws_size;
  const float* x_sh  = (const float*)d_in[0];
  const float* x_t1  = (const float*)d_in[1];
  const float* x_t2  = (const float*)d_in[2];
  const float* w1_sh = (const float*)d_in[3];
  const float* b1_sh = (const float*)d_in[4];
  const float* w2_sh = (const float*)d_in[5];
  const float* b2_sh = (const float*)d_in[6];
  const float* w1_t1 = (const float*)d_in[7];
  const float* b1_t1 = (const float*)d_in[8];
  const float* w2_t1 = (const float*)d_in[9];
  const float* b2_t1 = (const float*)d_in[10];
  const float* w1_t2 = (const float*)d_in[11];
  const float* b1_t2 = (const float*)d_in[12];
  const float* w2_t2 = (const float*)d_in[13];
  const float* b2_t2 = (const float*)d_in[14];
  const float* wg_sh = (const float*)d_in[15];
  const float* bg_sh = (const float*)d_in[16];
  const float* wg_t1 = (const float*)d_in[17];
  const float* bg_t1 = (const float*)d_in[18];
  const float* wg_t2 = (const float*)d_in[19];
  const float* bg_t2 = (const float*)d_in[20];
  float* out = (float*)d_out;

  // workspace carve-up (~101 MB total)
  char* ws = (char*)d_ws;
  unsigned short* xb  = (unsigned short*)ws;                 // [3][B][I] bf16
  unsigned short* w1T = xb  + (size_t)3 * B_ * I_;           // [12][H][I] bf16
  unsigned short* w2T = w1T + (size_t)12 * H_ * I_;          // [12][O][H] bf16
  float* gs = (float*)(w2T + (size_t)12 * O_ * H_);          // [B][12]
  float* g1 = gs + (size_t)B_ * 12;                          // [B][8]
  float* g2 = g1 + (size_t)B_ * 8;                           // [B][8]

  // 1) convert activations to bf16
  dim3 gcx(((size_t)B_ * I_ + 255) / 256, 3);
  cgc_cvt_x_kernel<<<gcx, 256, 0, stream>>>(x_sh, x_t1, x_t2, xb);

  // 2) transpose+convert weights: w1 [4,I,H]->[4,H,I], w2 [4,H,O]->[4,O,H]
  dim3 g1t(H_ / 32, I_ / 32, 4);
  cgc_transpose_bf16_kernel<<<g1t, 256, 0, stream>>>(w1_t1, w1T,                        I_, H_);
  cgc_transpose_bf16_kernel<<<g1t, 256, 0, stream>>>(w1_t2, w1T + (size_t)4 * H_ * I_,  I_, H_);
  cgc_transpose_bf16_kernel<<<g1t, 256, 0, stream>>>(w1_sh, w1T + (size_t)8 * H_ * I_,  I_, H_);
  dim3 g2t(O_ / 32, H_ / 32, 4);
  cgc_transpose_bf16_kernel<<<g2t, 256, 0, stream>>>(w2_t1, w2T,                        H_, O_);
  cgc_transpose_bf16_kernel<<<g2t, 256, 0, stream>>>(w2_t2, w2T + (size_t)4 * O_ * H_,  H_, O_);
  cgc_transpose_bf16_kernel<<<g2t, 256, 0, stream>>>(w2_sh, w2T + (size_t)8 * O_ * H_,  H_, O_);

  // 3) gates
  cgc_gate_kernel<<<B_, 128, 0, stream>>>(x_sh, x_t1, x_t2, wg_sh, bg_sh,
                                          wg_t1, bg_t1, wg_t2, bg_t2, gs, g1, g2);

  // 4) zero outputs (atomically accumulated)
  hipMemsetAsync(d_out, 0, (size_t)3 * B_ * O_ * sizeof(float), stream);

  // 5) fused experts + combine
  size_t lds = (size_t)16 * LDH * 2 + (size_t)3 * 16 * O_ * 4 + 3 * 16 * 4;
  hipFuncSetAttribute((const void*)cgc_expert_kernel,
                      hipFuncAttributeMaxDynamicSharedMemorySize, (int)lds);
  dim3 ge(B_ / 16, 3);
  cgc_expert_kernel<<<ge, 256, lds, stream>>>(xb, w1T, w2T,
                                              b1_t1, b1_t2, b1_sh,
                                              b2_t1, b2_t2, b2_sh,
                                              gs, g1, g2, out);
}
